// SelfAttention_48842368090381
// MI455X (gfx1250) — matrix operand order repaired and checked by reading, not compiled, not run on hardware
//
#include <hip/hip_runtime.h>
#include <hip/hip_bf16.h>

// ---------------------------------------------------------------------------
// CDNA5 (gfx1250) self-attention forward, bf16 WMMA with f32 accumulation.
// ---------------------------------------------------------------------------

typedef __attribute__((ext_vector_type(16))) __bf16 v16bf;
typedef __attribute__((ext_vector_type(8)))  float  v8f;
typedef int v4i __attribute__((vector_size(16)));

union Frag {
    v16bf    v;
    unsigned u[8];
};

__device__ __forceinline__ v8f wmma_bf16(const Frag& a, const Frag& b, v8f c) {
    // v_wmma_f32_16x16x32_bf16: D = A(16x32) * B(32x16) + C
    return __builtin_amdgcn_wmma_f32_16x16x32_bf16(
        /*neg_a=*/false, a.v, /*neg_b=*/false, b.v,
        /*c_mod=*/(short)0, c, /*reuse_a=*/false, /*reuse_b=*/false);
}

__device__ __forceinline__ v8f zero8() {
    v8f v;
#pragma unroll
    for (int i = 0; i < 8; ++i) v[i] = 0.0f;
    return v;
}

__device__ __forceinline__ unsigned short f32_to_bf16(float f) {
    unsigned u = __builtin_bit_cast(unsigned, f);
    u += 0x7fffu + ((u >> 16) & 1u);   // round-to-nearest-even
    return (unsigned short)(u >> 16);
}

// ---- optional gfx1250 async global->LDS copy (ASYNCcnt-tracked) -----------
#if defined(__HIP_DEVICE_COMPILE__) && \
    __has_builtin(__builtin_amdgcn_global_load_async_to_lds_b128)
#define ASYNC_LDS 1
__device__ __forceinline__ void async_copy16(const void* g, void* l) {
    // Signature (from compiler diagnostic): (v4i AS1*, v4i AS3*, imm, imm)
    __builtin_amdgcn_global_load_async_to_lds_b128(
        (__attribute__((address_space(1))) v4i*)(g),
        (__attribute__((address_space(3))) v4i*)(l),
        /*offset=*/0, /*cpol=*/0);
}
__device__ __forceinline__ void wait_async0() {
#if __has_builtin(__builtin_amdgcn_s_wait_asynccnt)
    __builtin_amdgcn_s_wait_asynccnt(0);
#else
    asm volatile("s_wait_asynccnt 0" ::: "memory");
#endif
}
#endif

// A-fragment (16x32 bf16) from row-major LDS tile, PITCH in u16 units.
// ISA layout: lanes 0-15 -> M=lane, K pairs {0,2,4,6} then {16,18,20,22};
// lanes 16-31 -> M=lane-16, K pairs +8.
template <int PITCH>
__device__ __forceinline__ void load_a(Frag& f, const unsigned short* base,
                                       int m, int hi) {
    const unsigned short* row = base + m * PITCH + hi * 8;
#pragma unroll
    for (int j = 0; j < 4; ++j)
        f.u[j] = *reinterpret_cast<const unsigned*>(row + 2 * j);
#pragma unroll
    for (int j = 0; j < 4; ++j)
        f.u[4 + j] = *reinterpret_cast<const unsigned*>(row + 16 + 2 * j);
}

// B-fragment (32x16 bf16) from row-major-[k][n] LDS tile.
// ISA layout: N = lane%16; lanes 0-15 K pairs {0,2,4,6} then {16,18,20,22};
// lanes 16-31 K pairs +8.
template <int PITCH>
__device__ __forceinline__ void load_b(Frag& f, const unsigned short* base,
                                       int krow, int n0) {
    const unsigned short* col = base + n0 + (krow & 15);
    const int kh = (krow >> 4) * 8;
#pragma unroll
    for (int j = 0; j < 8; ++j) {
        const int k = (j < 4) ? (kh + 2 * j) : (16 + kh + 2 * (j - 4));
        f.u[j] = (unsigned)col[k * PITCH] | ((unsigned)col[(k + 1) * PITCH] << 16);
    }
}

// ---------------------------------------------------------------------------
// Kernel 1: positional encoding. pe[s][2i] = sin(v[s]*e^{-i/128}),
//           pe[s][2i+1] = cos(...).  (log10(10000)=4 -> -4*2i/1024 = -i/128)
// ---------------------------------------------------------------------------
__global__ void pe_kernel(const float* __restrict__ vels,
                          float* __restrict__ pe) {
    const int s = blockIdx.x;
    const float v = vels[s];
    for (int i = threadIdx.x; i < 512; i += blockDim.x) {
        float dt = __expf(-(float)i * 0.0078125f);
        float ang = v * dt;
        float sn, cs;
        __sincosf(ang, &sn, &cs);
        pe[s * 1024 + 2 * i]     = sn;
        pe[s * 1024 + 2 * i + 1] = cs;
    }
}

// ---------------------------------------------------------------------------
// Kernel 2: QKV projection GEMM.  y[m][n] = sum_k x[m][k]*W[n][k] + b[n]
// M=16384 (b,s), N=1024, K=1024.  Block = 256 thr (8 waves), tile 128Mx128N,
// K-chunks of 32 staged in LDS as bf16 (W transposed to [k][n]).
// Output stored bf16 at [b][h][s][64].
// ---------------------------------------------------------------------------
__global__ __launch_bounds__(256)
void qkv_kernel(const float* __restrict__ x,
                const float* __restrict__ Wq, const float* __restrict__ bq,
                const float* __restrict__ Wk, const float* __restrict__ bk,
                const float* __restrict__ Wv, const float* __restrict__ bv,
                unsigned short* __restrict__ Qo,
                unsigned short* __restrict__ Ko,
                unsigned short* __restrict__ Vo) {
    __shared__ __align__(16) unsigned short Xsh[128 * 34];  // 128 m x 32 k (pad 34)
    __shared__ __align__(16) unsigned short Wt[32 * 130];   // 32 k x 128 n (pad 130)

    const int tid  = threadIdx.x;
    const int wave = tid >> 5;
    const int lane = tid & 31;
    const int m    = lane & 15;
    const int hi   = lane >> 4;

    const int nt = blockIdx.x;  // 0..7   (128-wide N tile)
    const int mt = blockIdx.y;  // 0..127 (128-wide M tile)
    const int z  = blockIdx.z;  // 0=q 1=k 2=v

    const float* W    = (z == 0) ? Wq : (z == 1) ? Wk : Wv;
    const float* bias = (z == 0) ? bq : (z == 1) ? bk : bv;
    unsigned short* dst = (z == 0) ? Qo : (z == 1) ? Ko : Vo;

    v8f acc[8];
#pragma unroll
    for (int j = 0; j < 8; ++j) acc[j] = zero8();

    for (int kb = 0; kb < 1024; kb += 32) {
        __syncthreads();
        // Stage X tile: thread -> row = tid/2, 16 floats at col (tid&1)*16.
        {
            const int row = tid >> 1;
            const int c0  = (tid & 1) * 16;
            const float* src = x + (size_t)(mt * 128 + row) * 1024 + kb + c0;
            if (kb + 32 < 1024) __builtin_prefetch(src + 32, 0, 3);
            unsigned short* drow = Xsh + row * 34 + c0;
#pragma unroll
            for (int i = 0; i < 4; ++i) {
                float4 f4 = reinterpret_cast<const float4*>(src)[i];
                drow[4 * i + 0] = f32_to_bf16(f4.x);
                drow[4 * i + 1] = f32_to_bf16(f4.y);
                drow[4 * i + 2] = f32_to_bf16(f4.z);
                drow[4 * i + 3] = f32_to_bf16(f4.w);
            }
        }
        // Stage W tile transposed: thread -> n = tid/2, 16 k at (tid&1)*16.
        {
            const int n  = tid >> 1;
            const int k0 = (tid & 1) * 16;
            const float* src = W + (size_t)(nt * 128 + n) * 1024 + kb + k0;
            if (kb + 32 < 1024) __builtin_prefetch(src + 32, 0, 3);
#pragma unroll
            for (int i = 0; i < 4; ++i) {
                float4 f4 = reinterpret_cast<const float4*>(src)[i];
                Wt[(k0 + 4 * i + 0) * 130 + n] = f32_to_bf16(f4.x);
                Wt[(k0 + 4 * i + 1) * 130 + n] = f32_to_bf16(f4.y);
                Wt[(k0 + 4 * i + 2) * 130 + n] = f32_to_bf16(f4.z);
                Wt[(k0 + 4 * i + 3) * 130 + n] = f32_to_bf16(f4.w);
            }
        }
        __syncthreads();

        Frag a;
        load_a<34>(a, Xsh + wave * 16 * 34, m, hi);
#pragma unroll
        for (int j = 0; j < 8; ++j) {
            Frag bfr;
            load_b<130>(bfr, Wt, lane, j * 16);
            acc[j] = wmma_bf16(a, bfr, acc[j]);
        }
    }

    // Store: C layout lane%16 = column, VGPR r = row r + 8*hi.
#pragma unroll
    for (int j = 0; j < 8; ++j) {
        const int ncol = nt * 128 + j * 16 + m;
        const float badd = bias[ncol];
        const int hh = ncol >> 6;
        const int dh = ncol & 63;
#pragma unroll
        for (int r = 0; r < 8; ++r) {
            const int row = mt * 128 + wave * 16 + 8 * hi + r;
            const int bb = row >> 10;
            const int ss = row & 1023;
            const size_t off = (((size_t)(bb * 16 + hh) * 1024) + ss) * 64 + dh;
            dst[off] = f32_to_bf16(acc[j][r] + badd);
        }
    }
}

// ---------------------------------------------------------------------------
// Kernel 3: flash-attention.  Block = 128 thr (4 waves), 64-query tile,
// streaming softmax over 32-key tiles.  Q/K/V are bf16 [b][h][s][64].
// scores = (Q K^T)/8 + pe[q][k];  out = softmax(scores) V  (f32 out).
// ---------------------------------------------------------------------------
__global__ __launch_bounds__(128)
void attn_kernel(const unsigned short* __restrict__ Q,
                 const unsigned short* __restrict__ K,
                 const unsigned short* __restrict__ V,
                 const float* __restrict__ pe,
                 float* __restrict__ out) {
    __shared__ __align__(16) unsigned short Qsh[64 * 72];    // 64 q x 64 d (pad 72)
    __shared__ __align__(16) unsigned short Kt [64 * 34];    // 64 d x 32 key (pad 34)
    __shared__ __align__(16) unsigned short Vsh[32 * 72];    // 32 key x 64 d (pad 72)
    __shared__ __align__(16) unsigned short Psh[4][16 * 34]; // per-wave P transpose

    const int tid  = threadIdx.x;
    const int wave = tid >> 5;
    const int lane = tid & 31;
    const int m    = lane & 15;
    const int hi   = lane >> 4;

    const int qt = blockIdx.x;  // 64-query tile
    const int h  = blockIdx.y;
    const int b  = blockIdx.z;

    const size_t bh = (size_t)(b * 16 + h) * (1024 * 64);
    const unsigned short* Qg = Q + bh;
    const unsigned short* Kg = K + bh;
    const unsigned short* Vg = V + bh;

    // Stage the 64x64 Q tile (bf16, row-major, pitch 72).
    {
        const int q  = tid >> 1;
        const int c2 = tid & 1;  // which 32-d half
        const unsigned short* src = Qg + (size_t)(qt * 64 + q) * 64 + c2 * 32;
        unsigned short* dstq = Qsh + q * 72 + c2 * 32;
#ifdef ASYNC_LDS
#pragma unroll
        for (int i = 0; i < 4; ++i) async_copy16(src + i * 8, dstq + i * 8);
        wait_async0();
#else
        uint4 r0 = reinterpret_cast<const uint4*>(src)[0];
        uint4 r1 = reinterpret_cast<const uint4*>(src)[1];
        uint4 r2 = reinterpret_cast<const uint4*>(src)[2];
        uint4 r3 = reinterpret_cast<const uint4*>(src)[3];
        unsigned* dq = reinterpret_cast<unsigned*>(dstq);
        dq[0] = r0.x;  dq[1] = r0.y;  dq[2]  = r0.z;  dq[3]  = r0.w;
        dq[4] = r1.x;  dq[5] = r1.y;  dq[6]  = r1.z;  dq[7]  = r1.w;
        dq[8] = r2.x;  dq[9] = r2.y;  dq[10] = r2.z;  dq[11] = r2.w;
        dq[12] = r3.x; dq[13] = r3.y; dq[14] = r3.z;  dq[15] = r3.w;
#endif
    }
    __syncthreads();

    Frag qf[2];
    load_a<72>(qf[0], Qsh + wave * 16 * 72, m, hi);        // d 0..31
    load_a<72>(qf[1], Qsh + wave * 16 * 72 + 32, m, hi);   // d 32..63

    float mrun[8], lrun[8];
#pragma unroll
    for (int r = 0; r < 8; ++r) { mrun[r] = -3.0e38f; lrun[r] = 0.0f; }
    v8f acc[4];
#pragma unroll
    for (int j = 0; j < 4; ++j) acc[j] = zero8();

    const int qg0 = qt * 64 + wave * 16 + 8 * hi;  // global row of VGPR 0

    for (int kt = 0; kt < 32; ++kt) {
        const int kb = kt * 32;
        __syncthreads();  // previous tile fully consumed
        // Stage K transposed [d][key] and V row-major [key][d].
        {
            const int key = tid >> 2;
            const int d0  = (tid & 3) * 16;
            const unsigned short* srcK = Kg + (size_t)(kb + key) * 64 + d0;
            const unsigned short* srcV = Vg + (size_t)(kb + key) * 64 + d0;
            if (kt + 1 < 32) {
                __builtin_prefetch(srcK + 2048, 0, 3);  // next 32-key tile
                __builtin_prefetch(srcV + 2048, 0, 3);
            }
            {
                uint4 a = reinterpret_cast<const uint4*>(srcK)[0];
                uint4 c = reinterpret_cast<const uint4*>(srcK)[1];
                unsigned w[8] = {a.x, a.y, a.z, a.w, c.x, c.y, c.z, c.w};
#pragma unroll
                for (int i = 0; i < 8; ++i) {
                    Kt[(d0 + 2 * i)     * 34 + key] = (unsigned short)(w[i] & 0xffffu);
                    Kt[(d0 + 2 * i + 1) * 34 + key] = (unsigned short)(w[i] >> 16);
                }
            }
#ifdef ASYNC_LDS
            {
                unsigned short* dstv = Vsh + key * 72 + d0;
#pragma unroll
                for (int i = 0; i < 2; ++i) async_copy16(srcV + i * 8, dstv + i * 8);
                wait_async0();
            }
#else
            {
                uint4 a = reinterpret_cast<const uint4*>(srcV)[0];
                uint4 c = reinterpret_cast<const uint4*>(srcV)[1];
                unsigned w[8] = {a.x, a.y, a.z, a.w, c.x, c.y, c.z, c.w};
                unsigned* dv = reinterpret_cast<unsigned*>(Vsh) + key * 36 + (tid & 3) * 8;
#pragma unroll
                for (int i = 0; i < 8; ++i) dv[i] = w[i];
            }
#endif
        }
        __syncthreads();

        // ---- scores: two 16x16 tiles over this 32-key block ----
        v8f a0 = zero8(), a1 = zero8();
#pragma unroll
        for (int c = 0; c < 2; ++c) {
            Frag b0, b1;
            load_b<34>(b0, Kt + c * 32 * 34, lane, 0);
            load_b<34>(b1, Kt + c * 32 * 34, lane, 16);
            a0 = wmma_bf16(qf[c], b0, a0);
            a1 = wmma_bf16(qf[c], b1, a1);
        }

        float s0[8], s1[8];
#pragma unroll
        for (int r = 0; r < 8; ++r) {
            const int qrow = qg0 + r;
            const float p0 = pe[(size_t)qrow * 1024 + kb + m];
            const float p1 = pe[(size_t)qrow * 1024 + kb + 16 + m];
            s0[r] = a0[r] * 0.125f + p0;
            s1[r] = a1[r] * 0.125f + p1;
        }

        // ---- streaming softmax (rows live in 16-lane halves) ----
#pragma unroll
        for (int r = 0; r < 8; ++r) {
            float nm = fmaxf(s0[r], s1[r]);
            nm = fmaxf(nm, __shfl_xor(nm, 1, 32));
            nm = fmaxf(nm, __shfl_xor(nm, 2, 32));
            nm = fmaxf(nm, __shfl_xor(nm, 4, 32));
            nm = fmaxf(nm, __shfl_xor(nm, 8, 32));
            nm = fmaxf(nm, mrun[r]);
            const float alpha = __expf(mrun[r] - nm);
            mrun[r] = nm;
            const float p0 = __expf(s0[r] - nm);
            const float p1 = __expf(s1[r] - nm);
            float rs = p0 + p1;
            rs += __shfl_xor(rs, 1, 32);
            rs += __shfl_xor(rs, 2, 32);
            rs += __shfl_xor(rs, 4, 32);
            rs += __shfl_xor(rs, 8, 32);
            lrun[r] = lrun[r] * alpha + rs;
            s0[r] = p0;
            s1[r] = p1;
#pragma unroll
            for (int j = 0; j < 4; ++j) acc[j][r] = acc[j][r] * alpha;
        }

        // ---- P (C layout) -> bf16 -> LDS -> A layout ----
        unsigned short* P = &Psh[wave][0];
#pragma unroll
        for (int r = 0; r < 8; ++r) {
            const int row = r + 8 * hi;
            P[row * 34 + m]      = f32_to_bf16(s0[r]);
            P[row * 34 + 16 + m] = f32_to_bf16(s1[r]);
        }
        // Same-wave LDS ops are processed in order; reads below see writes above.
        Frag pf;
        load_a<34>(pf, P, m, hi);

        // ---- out += P (16x32) x V (32x64) ----
#pragma unroll
        for (int j = 0; j < 4; ++j) {
            Frag vf;
            load_b<72>(vf, Vsh, lane, j * 16);
            acc[j] = wmma_bf16(pf, vf, acc[j]);
        }
    }

    // Finalize: divide by row sums, write (B,S,D) f32.
#pragma unroll
    for (int j = 0; j < 4; ++j) {
        const int col = h * 64 + j * 16 + m;
#pragma unroll
        for (int r = 0; r < 8; ++r) {
            const int qrow = qg0 + r;
            out[((size_t)b * 1024 + qrow) * 1024 + col] = acc[j][r] / lrun[r];
        }
    }
}

// ---------------------------------------------------------------------------
// Launch
// ---------------------------------------------------------------------------
extern "C" void kernel_launch(void* const* d_in, const int* in_sizes, int n_in,
                              void* d_out, int out_size, void* d_ws, size_t ws_size,
                              hipStream_t stream) {
    (void)in_sizes; (void)n_in; (void)out_size; (void)ws_size;

    const float* x    = (const float*)d_in[0];
    const float* vels = (const float*)d_in[1];
    const float* Wq   = (const float*)d_in[2];
    const float* bq   = (const float*)d_in[3];
    const float* Wk   = (const float*)d_in[4];
    const float* bk   = (const float*)d_in[5];
    const float* Wv   = (const float*)d_in[6];
    const float* bv   = (const float*)d_in[7];
    float* out = (float*)d_out;

    char* ws = (char*)d_ws;
    const size_t MB = (size_t)1 << 20;
    float*          pe = (float*)ws;                        //  4 MB
    unsigned short* Qb = (unsigned short*)(ws + 4 * MB);    // 32 MB
    unsigned short* Kb = (unsigned short*)(ws + 36 * MB);   // 32 MB
    unsigned short* Vb = (unsigned short*)(ws + 68 * MB);   // 32 MB

    pe_kernel<<<1024, 256, 0, stream>>>(vels, pe);
    qkv_kernel<<<dim3(8, 128, 3), 256, 0, stream>>>(x, Wq, bq, Wk, bk, Wv, bv,
                                                    Qb, Kb, Vb);
    attn_kernel<<<dim3(16, 16, 16), 128, 0, stream>>>(Qb, Kb, Vb, pe, out);
}
